// SelfAttention_42236708389264
// MI455X (gfx1250) — compile-verified, matrix operand order checked by reading, not run on hardware
//
#include <hip/hip_runtime.h>

// ---------------------------------------------------------------------------
// Types for WMMA fragments (wave32, V_WMMA_F32_16X16X32_F16)
// ---------------------------------------------------------------------------
typedef _Float16 h8   __attribute__((ext_vector_type(8)));
typedef _Float16 v16h __attribute__((ext_vector_type(16)));
typedef float    v8f  __attribute__((ext_vector_type(8)));
typedef float    f4   __attribute__((ext_vector_type(4)));

union V16H { v16h v; h8 h[2]; };

// Convert 8 consecutive f32 -> 8 f16 (two 16B loads + v_cvt)
__device__ __forceinline__ h8 cvt8(const float* p) {
  f4 a = *(const f4*)p;
  f4 b = *(const f4*)(p + 4);
  h8 r;
  r[0] = (_Float16)a[0]; r[1] = (_Float16)a[1];
  r[2] = (_Float16)a[2]; r[3] = (_Float16)a[3];
  r[4] = (_Float16)b[0]; r[5] = (_Float16)b[1];
  r[6] = (_Float16)b[2]; r[7] = (_Float16)b[3];
  return r;
}

// A-fragment (16x32 f16, MxK) from row-major f32 source. ISA layout:
// lane l holds row r0+(l&15); element i holds K = k0 + 16*(i>>3) + 8*(l>>4) + (i&7)
__device__ __forceinline__ v16h frag_f32(const float* base, int ld, int r0, int k0, int lane) {
  const float* p = base + (size_t)(r0 + (lane & 15)) * ld + k0 + ((lane >> 4) << 3);
  V16H u; u.h[0] = cvt8(p); u.h[1] = cvt8(p + 16);
  return u.v;
}

// Same layout from an f16 source (A-fragment, or B-fragment from a
// column-major/"transposed" matrix where lane's N-index selects the row).
__device__ __forceinline__ v16h frag_f16(const _Float16* base, int ld, int r0, int k0, int lane) {
  const _Float16* p = base + (size_t)(r0 + (lane & 15)) * ld + k0 + ((lane >> 4) << 3);
  V16H u; u.h[0] = *(const h8*)p; u.h[1] = *(const h8*)(p + 16);
  return u.v;
}

__device__ __forceinline__ v8f wmma32(v16h a, v16h b, v8f c) {
  return __builtin_amdgcn_wmma_f32_16x16x32_f16(false, a, false, b, (short)0, c, false, false);
}

// ---------------------------------------------------------------------------
// Kernel 1: weight convert + transpose: W[K][N] f32 -> Wt[N][K] f16
// ---------------------------------------------------------------------------
__global__ __launch_bounds__(256)
void wt_kernel(const float* __restrict__ W, _Float16* __restrict__ Wt, int K, int N) {
  int idx = blockIdx.x * 256 + threadIdx.x;
  if (idx >= K * N) return;
  int k = idx / N;
  int n = idx - k * N;
  Wt[(size_t)n * K + k] = (_Float16)W[idx];
}

// ---------------------------------------------------------------------------
// Kernel 2/5: WMMA GEMM  Y[M][N] = A[M][K] @ Wt[N][K]^T + bias
// block = 256 threads = 8 waves; block tile 128x64; wave tile 32x32
// ---------------------------------------------------------------------------
template<bool A_F32, bool OUT_F16>
__global__ __launch_bounds__(256)
void gemm_wmma_kernel(const void* __restrict__ Ap, const _Float16* __restrict__ Bt,
                      const float* __restrict__ bias, void* __restrict__ Yp,
                      int M, int N, int K) {
  const int lane = threadIdx.x & 31;
  const int wave = threadIdx.x >> 5;
  const int wm = wave >> 1;                 // 0..3
  const int wn = wave & 1;                  // 0..1
  const int m0 = blockIdx.x * 128 + wm * 32;
  const int n0 = blockIdx.y * 64  + wn * 32;

  v8f acc[2][2] = {};
  for (int k0 = 0; k0 < K; k0 += 32) {
    v16h a0, a1;
    if (A_F32) {
      const float* A = (const float*)Ap;
      a0 = frag_f32(A, K, m0,      k0, lane);
      a1 = frag_f32(A, K, m0 + 16, k0, lane);
    } else {
      const _Float16* A = (const _Float16*)Ap;
      a0 = frag_f16(A, K, m0,      k0, lane);
      a1 = frag_f16(A, K, m0 + 16, k0, lane);
    }
    v16h b0 = frag_f16(Bt, K, n0,      k0, lane);
    v16h b1 = frag_f16(Bt, K, n0 + 16, k0, lane);
    acc[0][0] = wmma32(a0, b0, acc[0][0]);
    acc[0][1] = wmma32(a0, b1, acc[0][1]);
    acc[1][0] = wmma32(a1, b0, acc[1][0]);
    acc[1][1] = wmma32(a1, b1, acc[1][1]);
  }

  // C/D layout: lane holds n = l&15; VGPR r holds m = r + 8*(l>>4)
  for (int im = 0; im < 2; im++)
    for (int jn = 0; jn < 2; jn++) {
      int n  = n0 + jn * 16 + (lane & 15);
      float bn = bias[n];
      int mb = m0 + im * 16 + ((lane >> 4) << 3);
      #pragma unroll
      for (int r = 0; r < 8; r++) {
        float vv = acc[im][jn][r] + bn;
        size_t off = (size_t)(mb + r) * N + n;
        if (OUT_F16) ((_Float16*)Yp)[off] = (_Float16)vv;
        else         ((float*)Yp)[off]    = vv;
      }
    }
}

// ---------------------------------------------------------------------------
// Kernel 3: L2-normalize q/k/v per (token, head); also emit V^T for attention.
// One wave per (token, slot): slot 0..7 = q head, 8..15 = k head, 16..23 = v head
// ---------------------------------------------------------------------------
__global__ __launch_bounds__(256)
void norm_kernel(_Float16* __restrict__ Q, _Float16* __restrict__ KV,
                 _Float16* __restrict__ Vt, int ntok) {
  int wid  = (blockIdx.x * 256 + threadIdx.x) >> 5;
  int lane = threadIdx.x & 31;
  int tok  = wid / 24;
  int slot = wid - tok * 24;
  if (tok >= ntok) return;

  _Float16* p;
  if (slot < 8)       p = Q  + (size_t)tok * 768  + slot * 96;
  else if (slot < 16) p = KV + (size_t)tok * 1536 + (slot - 8) * 96;
  else                p = KV + (size_t)tok * 1536 + 768 + (slot - 16) * 96;

  float x0 = (float)p[lane], x1 = (float)p[lane + 32], x2 = (float)p[lane + 64];
  float ss = x0 * x0 + x1 * x1 + x2 * x2;
  #pragma unroll
  for (int o = 16; o > 0; o >>= 1) ss += __shfl_xor(ss, o, 32);
  float sc = 1.0f / fmaxf(sqrtf(ss), 1e-12f);
  x0 *= sc; x1 *= sc; x2 *= sc;
  p[lane]      = (_Float16)x0;
  p[lane + 32] = (_Float16)x1;
  p[lane + 64] = (_Float16)x2;

  if (slot >= 16) {                          // also write V transposed: Vt[bth][hd][l]
    int h = slot - 16;
    size_t base = ((size_t)(tok >> 9) * 8 + h) * (size_t)(96 * 512) + (tok & 511);
    Vt[base + (size_t)lane        * 512] = (_Float16)x0;
    Vt[base + (size_t)(lane + 32) * 512] = (_Float16)x1;
    Vt[base + (size_t)(lane + 64) * 512] = (_Float16)x2;
  }
}

// ---------------------------------------------------------------------------
// Kernel 4: attention per (b,t,h) and q-tile of 32 rows.
// pass1: S = qk^T/sqrt(96) via WMMA -> LDS f16 (32x512 = 32KB)
// pass2: row softmax in LDS
// pass3: O = P @ V via WMMA (P A-frags from LDS, V^T B-frags from global)
// ---------------------------------------------------------------------------
#define QT 32
__global__ __launch_bounds__(256)
void attn_kernel(const _Float16* __restrict__ Q, const _Float16* __restrict__ KV,
                 const _Float16* __restrict__ Vt, _Float16* __restrict__ O) {
  __shared__ _Float16 S[QT][512];

  const int bth  = blockIdx.x;      // (b*T+t)*H + h
  const int qt   = blockIdx.y;      // 0..15
  const int h    = bth & 7;
  const int bt   = bth >> 3;
  const int lane = threadIdx.x & 31;
  const int wave = threadIdx.x >> 5;
  const int wq   = wave >> 2;       // 0..1 : q-row group of 16
  const int wk   = wave & 3;        // 0..3 : key quarter of 128

  const size_t tokbase = (size_t)bt * 512 + (size_t)qt * QT;
  const _Float16* qp = Q  + tokbase * 768 + h * 96;                    // ld 768
  const _Float16* kp = KV + (size_t)bt * 512 * 1536 + h * 96;          // ld 1536

  // ---- pass 1: scores ----
  v16h aq[3];
  #pragma unroll
  for (int s = 0; s < 3; s++) aq[s] = frag_f16(qp, 768, wq * 16, s * 32, lane);

  const float scale = 0.1020620726159658f;   // 1/sqrt(96)
  for (int kc = 0; kc < 8; kc++) {
    int key0 = wk * 128 + kc * 16;
    v8f acc = {};
    #pragma unroll
    for (int s = 0; s < 3; s++) {
      v16h bk = frag_f16(kp, 1536, key0, s * 32, lane);   // k rows are B-frag columns
      acc = wmma32(aq[s], bk, acc);
    }
    int n  = key0 + (lane & 15);
    int rb = wq * 16 + ((lane >> 4) << 3);
    #pragma unroll
    for (int r = 0; r < 8; r++) S[rb + r][n] = (_Float16)(acc[r] * scale);
  }
  __syncthreads();

  // ---- pass 2: softmax (8 lanes per row, 64 cols each) ----
  {
    int row = threadIdx.x >> 3;
    int c0  = (threadIdx.x & 7) * 64;
    float mx = -1e30f;
    for (int c = 0; c < 64; c++) mx = fmaxf(mx, (float)S[row][c0 + c]);
    #pragma unroll
    for (int o = 1; o < 8; o <<= 1) mx = fmaxf(mx, __shfl_xor(mx, o, 32));
    float sum = 0.f;
    for (int c = 0; c < 64; c++) {
      float e = __expf((float)S[row][c0 + c] - mx);
      sum += e;
      S[row][c0 + c] = (_Float16)e;
    }
    #pragma unroll
    for (int o = 1; o < 8; o <<= 1) sum += __shfl_xor(sum, o, 32);
    float inv = 1.0f / sum;
    for (int c = 0; c < 64; c++)
      S[row][c0 + c] = (_Float16)((float)S[row][c0 + c] * inv);
  }
  __syncthreads();

  // ---- pass 3: O = P @ V ----
  const _Float16* vtb = Vt + (size_t)bth * (size_t)(96 * 512);   // [hd][key], ld 512
  const int wc = wave & 3;
  const int nc = (wc < 2) ? 2 : 1;     // 6 hd-chunks of 16 spread over 4 col-waves
  v8f acc[2] = {};
  for (int kk = 0; kk < 512; kk += 32) {
    v16h ap = frag_f16(&S[wq * 16][0], 512, 0, kk, lane);        // ds_load_b128 x2
    for (int ci = 0; ci < nc; ci++) {
      int c = wc + ci * 4;
      v16h bv = frag_f16(vtb, 512, c * 16, kk, lane);
      acc[ci] = wmma32(ap, bv, acc[ci]);
    }
  }
  for (int ci = 0; ci < nc; ci++) {
    int c   = wc + ci * 4;
    int col = h * 96 + c * 16 + (lane & 15);
    int rb  = wq * 16 + ((lane >> 4) << 3);
    #pragma unroll
    for (int r = 0; r < 8; r++) {
      size_t tok = tokbase + rb + r;
      O[tok * 768 + col] = (_Float16)acc[ci][r];
    }
  }
}

// ---------------------------------------------------------------------------
// Host-side launch
// ---------------------------------------------------------------------------
extern "C" void kernel_launch(void* const* d_in, const int* in_sizes, int n_in,
                              void* d_out, int out_size, void* d_ws, size_t ws_size,
                              hipStream_t stream) {
  (void)in_sizes; (void)n_in; (void)out_size; (void)ws_size;
  const float* x   = (const float*)d_in[0];
  const float* Wq  = (const float*)d_in[1];
  const float* bq  = (const float*)d_in[2];
  const float* Wkv = (const float*)d_in[3];
  const float* bkv = (const float*)d_in[4];
  const float* Wm  = (const float*)d_in[5];
  const float* bm  = (const float*)d_in[6];

  const int M = 16384, C = 768;

  char* w = (char*)d_ws;
  _Float16* Wqt  = (_Float16*)w; w += (size_t)C * C * 2;
  _Float16* Wkvt = (_Float16*)w; w += (size_t)C * 2 * C * 2;
  _Float16* Wmt  = (_Float16*)w; w += (size_t)C * C * 2;
  _Float16* Qb   = (_Float16*)w; w += (size_t)M * C * 2;
  _Float16* KVb  = (_Float16*)w; w += (size_t)M * 2 * C * 2;
  _Float16* Vt   = (_Float16*)w; w += (size_t)256 * 96 * 512 * 2;
  _Float16* Ob   = (_Float16*)w;

  wt_kernel<<<(C * C + 255) / 256, 256, 0, stream>>>(Wq, Wqt, C, C);
  wt_kernel<<<(C * 2 * C + 255) / 256, 256, 0, stream>>>(Wkv, Wkvt, C, 2 * C);
  wt_kernel<<<(C * C + 255) / 256, 256, 0, stream>>>(Wm, Wmt, C, C);

  dim3 gq(M / 128, C / 64);
  gemm_wmma_kernel<true, true><<<gq, 256, 0, stream>>>(x, Wqt, bq, Qb, M, C, C);
  dim3 gkv(M / 128, (2 * C) / 64);
  gemm_wmma_kernel<true, true><<<gkv, 256, 0, stream>>>(x, Wkvt, bkv, KVb, M, 2 * C, C);

  norm_kernel<<<(M * 24 * 32) / 256, 256, 0, stream>>>(Qb, KVb, Vt, M);

  dim3 ga(256, 512 / QT);
  attn_kernel<<<ga, 256, 0, stream>>>(Qb, KVb, Vt, Ob);

  dim3 go(M / 128, C / 64);
  gemm_wmma_kernel<false, false><<<go, 256, 0, stream>>>(Ob, Wmt, bm, d_out, M, C, C);
}
